// GAT_26405458935809
// MI455X (gfx1250) — matrix operand order repaired and checked by reading, not compiled, not run on hardware
//
#include <hip/hip_runtime.h>
#include <hip/hip_bf16.h>

typedef __attribute__((ext_vector_type(16))) _Float16 v16h;
typedef __attribute__((ext_vector_type(8)))  _Float16 v8h;
typedef __attribute__((ext_vector_type(8)))  float    v8f;
typedef __attribute__((ext_vector_type(4)))  unsigned v4u;
typedef __attribute__((ext_vector_type(8)))  int      v8i;
typedef __attribute__((ext_vector_type(4)))  int      v4i;

#define N_NODES 8192
#define F_IN    512
#define F_OUT   256
#define NBT     272    // B cols: 256 (w*h) + 1 (den col = w) + 15 zero pad
#define NTILE   17     // NBT / 16
#define KSPLIT  4      // K-dimension split of the big GEMM (occupancy)
#define LDS_ROW 40     // halfs per B column in LDS: 32 data + 4 DWORD pad (80 B)
#define LDS_CHUNK (NBT * LDS_ROW)   // 10880 halfs = 21760 B per buffer

static __device__ __forceinline__ v8f wmma_f16(v16h a, v16h b, v8f c) {
    return __builtin_amdgcn_wmma_f32_16x16x32_f16(false, a, false, b, (short)0, c,
                                                  false, false);
}

// TDM: one DMA moves a 272-row x 64B tile of Bt (row stride 8192 halfs) into LDS,
// inserting 16B of pad after every 64B row (pad_interval=3 -> 16 DWORDs,
// pad_amount=3 -> 4 DWORDs) => LDS column stride 80B (bank-conflict-free b128 reads).
static __device__ __forceinline__ void tdm_issue(const _Float16* gsrc, _Float16* ldsdst) {
    unsigned long long ga = (unsigned long long)gsrc;
    unsigned lds = (unsigned)(unsigned long long)ldsdst;   // LDS aperture: addr[31:0]
    v4u g0;
    g0[0] = 1u;                                            // count=1, user mode
    g0[1] = lds;                                           // lds_addr
    g0[2] = (unsigned)(ga & 0xFFFFFFFFull);                // global_addr[31:0]
    g0[3] = (unsigned)((ga >> 32) & 0x1FFFFFFull) | (2u << 30);  // addr[56:32] | type=2
    v8i g1;
    g1[0] = (1 << 16) | (1 << 20) | (3 << 22) | (3 << 25); // data_size=2B, pad en/int/amt
    g1[1] = (int)(8192u << 16);                            // tensor_dim0 lo16 (bits 63:48)
    g1[2] = (int)((8192u >> 16) | (272u << 16));           // dim0 hi | tensor_dim1 lo
    g1[3] = (int)(32u << 16);                              // dim1 hi=0 | tile_dim0=32
    g1[4] = 272;                                           // tile_dim1=272, tile_dim2=0
    g1[5] = 8192;                                          // tensor_dim0_stride lo32
    g1[6] = 0;
    g1[7] = 0;
    v4i gz  = {0, 0, 0, 0};                                // 2-D tensor: groups 2/3 unused
    v8i gz8 = {0, 0, 0, 0, 0, 0, 0, 0};
    __builtin_amdgcn_tensor_load_to_lds(g0, g1, gz, gz, gz8, 0);
}

// ---------------- kernel 0: W (512x256 f32) -> Wt (256x512 f16) ---------------
__global__ void k_wt(const float* __restrict__ W, _Float16* __restrict__ Wt) {
    int idx = blockIdx.x * 256 + threadIdx.x;
    int c = idx >> 9;
    int k = idx & 511;
    Wt[(size_t)c * F_IN + k] = (_Float16)W[(size_t)k * F_OUT + c];
}

// ---------------- kernel 1: h = x @ W + bias (f32 out), B double-buffered -----
__global__ void __launch_bounds__(256) k_h(const float* __restrict__ x,
                                           const _Float16* __restrict__ Wt,
                                           const float* __restrict__ bias,
                                           float* __restrict__ h) {
    const int lane = threadIdx.x & 31;
    const int wv   = threadIdx.x >> 5;
    const int ln   = lane & 15;
    const int lh   = lane >> 4;
    const int row_base = (blockIdx.x * 8 + wv) * 16;

    v8f zero = {0.f,0.f,0.f,0.f,0.f,0.f,0.f,0.f};
    v8f acc[16];
    #pragma unroll
    for (int t = 0; t < 16; ++t) acc[t] = zero;

    const float* ap = x + (size_t)(row_base + ln) * F_IN;
    v16h bc = *(const v16h*)(Wt + (size_t)ln * F_IN + lh * 16);
    for (int kc = 0; kc < F_IN; kc += 32) {
        const int ka = kc + lh * 16;
        float4 q0 = *(const float4*)(ap + ka);
        float4 q1 = *(const float4*)(ap + ka + 4);
        float4 q2 = *(const float4*)(ap + ka + 8);
        float4 q3 = *(const float4*)(ap + ka + 12);
        v16h a;
        a[0]=(_Float16)q0.x;  a[1]=(_Float16)q0.y;  a[2]=(_Float16)q0.z;  a[3]=(_Float16)q0.w;
        a[4]=(_Float16)q1.x;  a[5]=(_Float16)q1.y;  a[6]=(_Float16)q1.z;  a[7]=(_Float16)q1.w;
        a[8]=(_Float16)q2.x;  a[9]=(_Float16)q2.y;  a[10]=(_Float16)q2.z; a[11]=(_Float16)q2.w;
        a[12]=(_Float16)q3.x; a[13]=(_Float16)q3.y; a[14]=(_Float16)q3.z; a[15]=(_Float16)q3.w;

        const int kcn = (kc + 32 < F_IN) ? kc + 32 : 0;
        const int kb  = kc  + lh * 16;
        const int kbn = kcn + lh * 16;
        #pragma unroll
        for (int t = 0; t < 16; ++t) {
            v16h bn;
            if (t < 15) bn = *(const v16h*)(Wt + (size_t)((t + 1) * 16 + ln) * F_IN + kb);
            else        bn = *(const v16h*)(Wt + (size_t)ln * F_IN + kbn);
            acc[t] = wmma_f16(a, bc, acc[t]);
            bc = bn;
        }
    }
    #pragma unroll
    for (int t = 0; t < 16; ++t) {
        int col = t * 16 + ln;
        float bv = bias[col];
        #pragma unroll
        for (int r = 0; r < 8; ++r) {
            int row = row_base + r + lh * 8;
            h[(size_t)row * F_OUT + col] = acc[t][r] + bv;
        }
    }
}

// ---------------- kernel 2: s_dst[j] = h[j,:] . phi_dst -----------------------
__global__ void __launch_bounds__(256) k_sdst(const float* __restrict__ h,
                                              const float* __restrict__ phi,
                                              float* __restrict__ sdst) {
    const int lane = threadIdx.x & 31;
    const int wv   = threadIdx.x >> 5;
    const int row  = blockIdx.x * 8 + wv;
    const float* hp = h + (size_t)row * F_OUT;
    const float* pd = phi + F_OUT;
    float s = 0.f;
    #pragma unroll
    for (int t = 0; t < 8; ++t) { int c = lane + t * 32; s += hp[c] * pd[c]; }
    #pragma unroll
    for (int off = 16; off >= 1; off >>= 1) s += __shfl_xor(s, off, 32);
    if (lane == 0) sdst[row] = s;
}

// ---------------- kernel 3: global max of s_dst -------------------------------
__global__ void k_max(const float* __restrict__ sdst, float* __restrict__ smax) {
    __shared__ float red[256];
    float m = -3.0e38f;
    for (int i = threadIdx.x; i < N_NODES; i += 256) m = fmaxf(m, sdst[i]);
    red[threadIdx.x] = m;
    __syncthreads();
    for (int s = 128; s > 0; s >>= 1) {
        if (threadIdx.x < s) red[threadIdx.x] = fmaxf(red[threadIdx.x], red[threadIdx.x + s]);
        __syncthreads();
    }
    if (threadIdx.x == 0) smax[0] = red[0];
}

// ---------------- kernel 4: w[j] = exp(s_dst[j] - smax) -----------------------
__global__ void k_w(const float* __restrict__ sdst, const float* __restrict__ smax,
                    float* __restrict__ w) {
    int j = blockIdx.x * 256 + threadIdx.x;
    w[j] = __expf(sdst[j] - smax[0]);
}

// ---------------- kernel 5: build Bt (272 x 8192, f16, column-major) ----------
__global__ void k_bt(const float* __restrict__ h, const float* __restrict__ w,
                     _Float16* __restrict__ Bt) {
    int j = blockIdx.x * 256 + threadIdx.x;
    int c = blockIdx.y;
    _Float16 v;
    if (c < F_OUT)        v = (_Float16)(w[j] * h[(size_t)j * F_OUT + c]);
    else if (c == F_OUT)  v = (_Float16)w[j];
    else                  v = (_Float16)0.f;
    Bt[(size_t)c * N_NODES + j] = v;
}

// ---------------- kernel 6: zero the num accumulator --------------------------
__global__ void k_zero(float* __restrict__ num) {
    int idx = blockIdx.x * 256 + threadIdx.x;
    if (idx < N_NODES * NBT) num[idx] = 0.f;
}

// ---------------- kernel 7: num += M_seg @ Bt_seg -----------------------------
// B chunk (272 cols x 32 K, 17KB) staged in LDS by the Tensor Data Mover, double
// buffered (TENSORcnt + barrier per chunk). adj int4 stream double-buffered in
// registers. Mask A-fragment (adj|I) built in-register, f16 WMMA accumulate.
__global__ void __launch_bounds__(256) k_main(const int* __restrict__ adj,
                                              const _Float16* __restrict__ Bt,
                                              float* __restrict__ num) {
    __shared__ __align__(128) _Float16 sbuf[2 * LDS_CHUNK];   // 43,520 B

    const int lane = threadIdx.x & 31;
    const int wv   = threadIdx.x >> 5;
    const int ln   = lane & 15;
    const int lh   = lane >> 4;
    const int row_base = (blockIdx.x * 8 + wv) * 16;
    const int rowg = row_base + ln;
    const int* ap = adj + (size_t)rowg * N_NODES;

    const int KSEG = N_NODES / KSPLIT;
    const int k0 = blockIdx.y * KSEG;
    const int k1 = k0 + KSEG;

    v8f zero = {0.f,0.f,0.f,0.f,0.f,0.f,0.f,0.f};
    v8f acc[NTILE];
    #pragma unroll
    for (int t = 0; t < NTILE; ++t) acc[t] = zero;

    // --- prologue: TDM-fill buffer 0; preload first adj chunk ---
    if (wv == 0) tdm_issue(Bt + k0, &sbuf[0]);
    int ka0 = k0 + lh * 8;
    int4 q0 = *(const int4*)(ap + ka0);
    int4 q1 = *(const int4*)(ap + ka0 + 4);
    int4 q2 = *(const int4*)(ap + ka0 + 16);
    int4 q3 = *(const int4*)(ap + ka0 + 20);
    if (wv == 0) __builtin_amdgcn_s_wait_tensorcnt(0);
    __syncthreads();

    int parity = 0;
    for (int kc = k0; kc < k1; kc += 32) {
        // kick off DMA for the next chunk into the other buffer
        if (wv == 0 && kc + 32 < k1) tdm_issue(Bt + kc + 32, &sbuf[(parity ^ 1) * LDS_CHUNK]);

        const int ka = kc + lh * 8;
        if (kc + 1024 < N_NODES) __builtin_prefetch(ap + ka + 1024, 0, 1);

        // build A fragment (mask | diagonal) from previously-loaded ints
        int vals[16] = { q0.x,q0.y,q0.z,q0.w, q1.x,q1.y,q1.z,q1.w,
                         q2.x,q2.y,q2.z,q2.w, q3.x,q3.y,q3.z,q3.w };
        v16h a;
        #pragma unroll
        for (int i = 0; i < 16; ++i) {
            int kg = ka + (i < 8 ? i : i + 8);
            a[i] = (vals[i] != 0 || kg == rowg) ? (_Float16)1.0f : (_Float16)0.0f;
        }

        // issue next adj chunk loads (wrap to k0 on last iter: always valid)
        const int kcn = (kc + 32 < k1) ? kc + 32 : k0;
        const int kan = kcn + lh * 8;
        int4 p0 = *(const int4*)(ap + kan);
        int4 p1 = *(const int4*)(ap + kan + 4);
        int4 p2 = *(const int4*)(ap + kan + 16);
        int4 p3 = *(const int4*)(ap + kan + 20);

        // 17 tiles: B fragments from LDS (80B column stride, conflict-free)
        const _Float16* bp = &sbuf[parity * LDS_CHUNK];
        #pragma unroll
        for (int t = 0; t < NTILE; ++t) {
            const _Float16* p = bp + (t * 16 + ln) * LDS_ROW + lh * 8;
            v8h lo = *(const v8h*)p;
            v8h hi = *(const v8h*)(p + 16);
            v16h b = __builtin_shufflevector(lo, hi, 0,1,2,3,4,5,6,7,
                                                     8,9,10,11,12,13,14,15);
            acc[t] = wmma_f16(a, b, acc[t]);
        }

        if (wv == 0) __builtin_amdgcn_s_wait_tensorcnt(0);
        __syncthreads();
        parity ^= 1;
        q0 = p0; q1 = p1; q2 = p2; q3 = p3;
    }

    // --- combine partial sums (f32 hardware atomics, no return) ---
    #pragma unroll
    for (int t = 0; t < NTILE; ++t) {
        int col = t * 16 + ln;
        #pragma unroll
        for (int r = 0; r < 8; ++r) {
            int row = row_base + r + lh * 8;
            unsafeAtomicAdd(&num[(size_t)row * NBT + col], acc[t][r]);
        }
    }
}

// ---------------- kernel 8: out = relu(num[:, :256] / num[:, 256]) ------------
__global__ void k_fin(const float* __restrict__ num, float* __restrict__ out) {
    int row = blockIdx.x;
    int c   = threadIdx.x;
    float den = num[(size_t)row * NBT + F_OUT];
    float v = num[(size_t)row * NBT + c] / den;
    out[(size_t)row * F_OUT + c] = v > 0.f ? v : 0.f;
}

// ---------------- launch ------------------------------------------------------
extern "C" void kernel_launch(void* const* d_in, const int* in_sizes, int n_in,
                              void* d_out, int out_size, void* d_ws, size_t ws_size,
                              hipStream_t stream) {
    const int*   adj  = (const int*)d_in[0];
    const float* x    = (const float*)d_in[1];
    const float* W    = (const float*)d_in[2];
    const float* bias = (const float*)d_in[3];
    const float* phi  = (const float*)d_in[4];
    float* out = (float*)d_out;

    char* ws = (char*)d_ws;
    float*    h    = (float*)   (ws + 0);          //  8,388,608 B
    float*    num  = (float*)   (ws + 8388608);    //  8,912,896 B
    _Float16* Bt   = (_Float16*)(ws + 17301504);   //  4,456,448 B
    _Float16* Wt   = (_Float16*)(ws + 21757952);   //    262,144 B
    float*    sdst = (float*)   (ws + 22020096);   //     32,768 B
    float*    wbuf = (float*)   (ws + 22052864);   //     32,768 B
    float*    smax = (float*)   (ws + 22085632);   //        256 B

    k_wt  <<<512, 256, 0, stream>>>(W, Wt);
    k_h   <<<64, 256, 0, stream>>>(x, Wt, bias, h);
    k_sdst<<<1024, 256, 0, stream>>>(h, phi, sdst);
    k_max <<<1, 256, 0, stream>>>(sdst, smax);
    k_w   <<<32, 256, 0, stream>>>(sdst, smax, wbuf);
    k_bt  <<<dim3(32, NBT), 256, 0, stream>>>(h, wbuf, Bt);
    k_zero<<<(N_NODES * NBT + 255) / 256, 256, 0, stream>>>(num);
    k_main<<<dim3(64, KSPLIT), 256, 0, stream>>>(adj, Bt, num);
    k_fin <<<N_NODES, 256, 0, stream>>>(num, out);
}
